// SPARSE_14001593385713
// MI455X (gfx1250) — matrix operand order repaired and checked by reading, not compiled, not run on hardware
//
#include <hip/hip_runtime.h>
#include <math.h>

typedef __attribute__((ext_vector_type(16))) _Float16 v16h;
typedef __attribute__((ext_vector_type(8)))  _Float16 v8h;
typedef __attribute__((ext_vector_type(4)))  _Float16 v4h;
typedef __attribute__((ext_vector_type(2)))  _Float16 v2h;
typedef __attribute__((ext_vector_type(8)))  float    v8f;

#define B_ROWS 4096
#define D_DIM  4096
#define H_DIM  1024

#define MBLK 128
#define NBLK 128
#define KBLK 32
#define LDST 56            // padded LDS row stride in halves (112B, 16B-aligned frag reads)
#define ABUF (MBLK * LDST) // halves per A buffer
#define BBUF (NBLK * LDST) // halves per B buffer

// d_out element offsets: sparse_x, mask, sparsity, actual_sparsity, l1
#define O_SPARSEX  ((size_t)0)
#define O_MASK     ((size_t)B_ROWS * D_DIM)
#define O_SPARSITY ((size_t)2 * B_ROWS * D_DIM)
#define O_ASPARS   ((size_t)2 * B_ROWS * D_DIM + B_ROWS)
#define O_L1       ((size_t)2 * B_ROWS * D_DIM + 2 * B_ROWS)

// ws byte offsets
#define WS_PARTIAL 0                                   // 32*4096 floats
#define WS_KBUF    (32 * B_ROWS * sizeof(float))       // 4096 ints
#define WS_L1ROW   (WS_KBUF + B_ROWS * sizeof(int))    // 4096 floats

// ---------------- Kernel 1: fused x@W1 -> relu -> (.)@W2 partial dots -------
// Register-prefetch + double-buffered LDS: global loads for tile k+1 overlap
// the WMMAs of tile k; one barrier per K-step.
__global__ __launch_bounds__(256)
void SPARSE_gemm_gate(const float* __restrict__ x, const float* __restrict__ W1,
                      const float* __restrict__ b1, const float* __restrict__ W2,
                      float* __restrict__ partial) {
  __shared__ __align__(16) _Float16 As[2 * ABUF];
  __shared__ __align__(16) _Float16 Bs[2 * BBUF];

  const int t    = threadIdx.x;
  const int lane = t & 31;
  const int wave = t >> 5;     // 0..7
  const int wm   = wave >> 2;  // 0..1 -> 64 rows each
  const int wn   = wave & 3;   // 0..3 -> 32 cols each
  const int m0   = blockIdx.y * MBLK;
  const int n0   = blockIdx.x * NBLK;
  const int lrow = lane & 15;
  const int half = (lane < 16) ? 0 : 1;

  v8f acc[4][2];
  #pragma unroll
  for (int i = 0; i < 4; ++i)
    #pragma unroll
    for (int j = 0; j < 2; ++j) acc[i][j] = v8f{};

  float4 aReg[4];    // A: 4 float4 chunks per thread
  float  bReg[8][2]; // B: 8 K-pairs per thread

  auto fetch = [&](int k0) {
    #pragma unroll
    for (int i = 0; i < 4; ++i) {
      int c   = t + 256 * i;   // float4 chunk id, 0..1023
      int row = c >> 3;
      int c4  = c & 7;
      aReg[i] = *(const float4*)(x + (size_t)(m0 + row) * D_DIM + k0 + c4 * 4);
    }
    #pragma unroll
    for (int i = 0; i < 8; ++i) {
      int p2 = t + 256 * i;        // K-pair id, 0..2047
      int n  = p2 & 127;
      int kk = (p2 >> 7) * 2;      // 0,2,..,30
      const float* gp = W1 + (size_t)(k0 + kk) * H_DIM + (n0 + n);
      bReg[i][0] = gp[0];
      bReg[i][1] = gp[H_DIM];
    }
  };

  auto stage = [&](int buf) {
    _Float16* Ab = As + buf * ABUF;
    _Float16* Bb = Bs + buf * BBUF;
    #pragma unroll
    for (int i = 0; i < 4; ++i) {
      int c   = t + 256 * i;
      int row = c >> 3;
      int c4  = c & 7;
      v4h h4 = { (_Float16)aReg[i].x, (_Float16)aReg[i].y,
                 (_Float16)aReg[i].z, (_Float16)aReg[i].w };
      *(v4h*)(&Ab[row * LDST + c4 * 4]) = h4;
    }
    #pragma unroll
    for (int i = 0; i < 8; ++i) {
      int p2 = t + 256 * i;
      int n  = p2 & 127;
      int kk = (p2 >> 7) * 2;
      v2h hp = { (_Float16)bReg[i][0], (_Float16)bReg[i][1] };
      *(v2h*)(&Bb[n * LDST + kk]) = hp;
    }
  };

  fetch(0);
  int buf = 0;
  for (int k0 = 0; k0 < D_DIM; k0 += KBLK) {
    stage(buf);
    if (k0 + KBLK < D_DIM) fetch(k0 + KBLK);  // overlaps with compute below
    __syncthreads();

    const _Float16* Ab = As + buf * ABUF;
    const _Float16* Bb = Bs + buf * BBUF;
    // A fragments: lanes<16 hold K {0..7,16..23}, lanes>=16 hold K {8..15,24..31}
    v16h afr[4];
    #pragma unroll
    for (int im = 0; im < 4; ++im) {
      int row  = wm * 64 + im * 16 + lrow;
      int koff = half ? 8 : 0;
      v8h lo = *(const v8h*)(&Ab[row * LDST + koff]);
      v8h hi = *(const v8h*)(&Ab[row * LDST + koff + 16]);
      afr[im] = __builtin_shufflevector(lo, hi, 0,1,2,3,4,5,6,7,8,9,10,11,12,13,14,15);
    }
    // B fragments: lanes<16 hold K {0..7,16..23}, lanes>=16 hold K {8..15,24..31} (N = lane%16)
    v16h bfr[2];
    #pragma unroll
    for (int jn = 0; jn < 2; ++jn) {
      int col  = wn * 32 + jn * 16 + lrow;
      int koff = half ? 8 : 0;
      v8h lo = *(const v8h*)(&Bb[col * LDST + koff]);
      v8h hi = *(const v8h*)(&Bb[col * LDST + koff + 16]);
      bfr[jn] = __builtin_shufflevector(lo, hi, 0,1,2,3,4,5,6,7,8,9,10,11,12,13,14,15);
    }
    #pragma unroll
    for (int im = 0; im < 4; ++im)
      #pragma unroll
      for (int jn = 0; jn < 2; ++jn)
        acc[im][jn] = __builtin_amdgcn_wmma_f32_16x16x32_f16(
            false, afr[im], false, bfr[jn], (short)0, acc[im][jn], false, false);
    buf ^= 1;
  }

  // Epilogue: relu(acc+b1) dot W2 over this wave's 32 columns, reduce 16 lanes,
  // write deterministic per-(nblock,wn) partial slot (no atomics).
  float b1v[2], w2v[2];
  #pragma unroll
  for (int jn = 0; jn < 2; ++jn) {
    int col = n0 + wn * 32 + jn * 16 + lrow;
    b1v[jn] = b1[col];
    w2v[jn] = W2[col];
  }
  const int slot = blockIdx.x * 4 + wn;  // 0..31
  #pragma unroll
  for (int im = 0; im < 4; ++im) {
    #pragma unroll
    for (int r = 0; r < 8; ++r) {
      float p = 0.f;
      #pragma unroll
      for (int jn = 0; jn < 2; ++jn) {
        float h = acc[im][jn][r] + b1v[jn];
        h = h > 0.f ? h : 0.f;
        p += h * w2v[jn];
      }
      p += __shfl_xor(p, 1, 32);
      p += __shfl_xor(p, 2, 32);
      p += __shfl_xor(p, 4, 32);
      p += __shfl_xor(p, 8, 32);
      if (lrow == 0) {
        int row = m0 + wm * 64 + im * 16 + r + (half ? 8 : 0);
        partial[(size_t)slot * B_ROWS + row] = p;
      }
    }
  }
}

// ------------- Kernel 2: reduce partials -> sigmoid gate -> sparsity, k -----
__global__ void SPARSE_gate(const float* __restrict__ partial, const float* __restrict__ b2,
                            float* __restrict__ out_sparsity, int* __restrict__ kbuf) {
  int m = blockIdx.x * blockDim.x + threadIdx.x;
  if (m >= B_ROWS) return;
  float z = 0.f;
  #pragma unroll 8
  for (int s = 0; s < 32; ++s) z += partial[(size_t)s * B_ROWS + m];
  z += b2[0];
  float sig = 1.f / (1.f + expf(-z));
  float sp  = 0.05f + 0.25f * sig;
  out_sparsity[m] = sp;
  float kf = rintf((float)D_DIM * (1.f - sp));  // RNE matches jnp.round
  int k = (int)kf;
  if (k < 1) k = 1;
  if (k > D_DIM) k = D_DIM;
  kbuf[m] = k;
}

// ------- Kernel 3: per-row radix select of k-th smallest |x|, outputs -------
__global__ __launch_bounds__(256)
void SPARSE_select(const float* __restrict__ x, const int* __restrict__ kbuf,
                   float* __restrict__ out, float* __restrict__ l1row) {
  __shared__ unsigned cache[D_DIM];     // abs bit patterns for this row (16KB)
  __shared__ unsigned hist[256];
  __shared__ unsigned sh_prefix, sh_k;
  __shared__ float rc[8], rl[8];

  const int row = blockIdx.x;
  const int t   = threadIdx.x;
  const float* xr = x + (size_t)row * D_DIM;

  // vectorized abs-bits cache fill (b128 loads)
  #pragma unroll
  for (int i = t; i < D_DIM / 4; i += 256) {
    float4 v = ((const float4*)xr)[i];
    cache[4 * i + 0] = __float_as_uint(v.x) & 0x7fffffffu;
    cache[4 * i + 1] = __float_as_uint(v.y) & 0x7fffffffu;
    cache[4 * i + 2] = __float_as_uint(v.z) & 0x7fffffffu;
    cache[4 * i + 3] = __float_as_uint(v.w) & 0x7fffffffu;
  }
  if (t == 0) { sh_prefix = 0u; sh_k = (unsigned)kbuf[row]; }
  __syncthreads();

  // 4 passes, MSB byte -> LSB byte; nonneg floats order like their uint bits
  for (int pass = 0; pass < 4; ++pass) {
    const int shift = 24 - 8 * pass;
    hist[t] = 0;
    __syncthreads();
    unsigned pfx = sh_prefix;
    for (int i = t; i < D_DIM; i += 256) {
      unsigned u = cache[i];
      bool match = (pass == 0) || ((u >> (shift + 8)) == (pfx >> (shift + 8)));
      if (match) atomicAdd(&hist[(u >> shift) & 255u], 1u);
    }
    __syncthreads();
    if (t == 0) {
      unsigned kk = sh_k, c = 0;
      int b = 0;
      for (; b < 256; ++b) {
        unsigned nc = c + hist[b];
        if (nc >= kk) { sh_k = kk - c; break; }
        c = nc;
      }
      sh_prefix = pfx | ((unsigned)b << shift);
    }
    __syncthreads();
  }

  const float thr = __uint_as_float(sh_prefix);  // exact k-th smallest |x|
  float l1 = 0.f, cnt = 0.f;
  float4* outS = (float4*)(out + O_SPARSEX + (size_t)row * D_DIM);
  float4* outM = (float4*)(out + O_MASK    + (size_t)row * D_DIM);
  #pragma unroll
  for (int i = t; i < D_DIM / 4; i += 256) {
    float4 xv = ((const float4*)xr)[i];
    float4 mk, sx;
    mk.x = (fabsf(xv.x) > thr) ? 1.f : 0.f;  sx.x = xv.x * mk.x;
    mk.y = (fabsf(xv.y) > thr) ? 1.f : 0.f;  sx.y = xv.y * mk.y;
    mk.z = (fabsf(xv.z) > thr) ? 1.f : 0.f;  sx.z = xv.z * mk.z;
    mk.w = (fabsf(xv.w) > thr) ? 1.f : 0.f;  sx.w = xv.w * mk.w;
    outS[i] = sx;
    outM[i] = mk;
    cnt += mk.x + mk.y + mk.z + mk.w;
    l1  += fabsf(sx.x) + fabsf(sx.y) + fabsf(sx.z) + fabsf(sx.w);
  }
  #pragma unroll
  for (int off = 16; off > 0; off >>= 1) {
    cnt += __shfl_xor(cnt, off, 32);
    l1  += __shfl_xor(l1,  off, 32);
  }
  if ((t & 31) == 0) { rc[t >> 5] = cnt; rl[t >> 5] = l1; }
  __syncthreads();
  if (t == 0) {
    float C = 0.f, L = 0.f;
    #pragma unroll
    for (int w = 0; w < 8; ++w) { C += rc[w]; L += rl[w]; }
    out[O_ASPARS + row] = C * (1.f / (float)D_DIM);
    l1row[row] = L;
  }
}

// ------------- Kernel 4: deterministic l1 mean reduction --------------------
__global__ void SPARSE_l1reduce(const float* __restrict__ l1row, float* __restrict__ out) {
  __shared__ float red[256];
  int t = threadIdx.x;
  float s = 0.f;
  for (int i = t; i < B_ROWS; i += 256) s += l1row[i];
  red[t] = s;
  __syncthreads();
  for (int off = 128; off > 0; off >>= 1) {
    if (t < off) red[t] += red[t + off];
    __syncthreads();
  }
  if (t == 0) out[O_L1] = red[0] * (1.f / (float)B_ROWS);
}

extern "C" void kernel_launch(void* const* d_in, const int* in_sizes, int n_in,
                              void* d_out, int out_size, void* d_ws, size_t ws_size,
                              hipStream_t stream) {
  const float* x  = (const float*)d_in[0];
  const float* W1 = (const float*)d_in[1];
  const float* b1 = (const float*)d_in[2];
  const float* W2 = (const float*)d_in[3];
  const float* b2 = (const float*)d_in[4];
  float* out = (float*)d_out;

  float* partial = (float*)((char*)d_ws + WS_PARTIAL);
  int*   kbuf    = (int*)  ((char*)d_ws + WS_KBUF);
  float* l1row   = (float*)((char*)d_ws + WS_L1ROW);

  dim3 g(H_DIM / NBLK, B_ROWS / MBLK);  // (8, 32)
  SPARSE_gemm_gate<<<g, 256, 0, stream>>>(x, W1, b1, W2, partial);
  SPARSE_gate<<<(B_ROWS + 255) / 256, 256, 0, stream>>>(partial, b2,
                                                        out + O_SPARSITY, kbuf);
  SPARSE_select<<<B_ROWS, 256, 0, stream>>>(x, kbuf, out, l1row);
  SPARSE_l1reduce<<<1, 256, 0, stream>>>(l1row, out);
}
